// RefinerMSLoss_16071767621695
// MI455X (gfx1250) — compile-verified, matrix operand order checked by reading, not run on hardware
//
#include <hip/hip_runtime.h>
#include <hip/hip_bf16.h>

typedef __bf16 bf16;
typedef __attribute__((ext_vector_type(16))) bf16  v16bf;
typedef __attribute__((ext_vector_type(8)))  bf16  v8bf;
typedef __attribute__((ext_vector_type(8)))  float v8f;

#define ALPHA 50.0f
#define BETA  2.0f
#define BASE  0.5f
#define MARGIN 0.1f
#define MSL_EPS 1e-16f
#define D_DIM 256

// ---------------------------------------------------------------------------
// Kernel 0: split f32 inputs into bf16 hi + bf16 lo residual planes.
// ---------------------------------------------------------------------------
__global__ __launch_bounds__(256) void msl_convert(
    const float* __restrict__ s, const float* __restrict__ t,
    bf16* __restrict__ sh, bf16* __restrict__ sl,
    bf16* __restrict__ th, bf16* __restrict__ tl, int total)
{
    int i = blockIdx.x * 256 + threadIdx.x;
    if (i >= total) return;
    float x = s[i];
    bf16 xh = (bf16)x;
    sh[i] = xh;
    sl[i] = (bf16)(x - (float)xh);
    float y = t[i];
    bf16 yh = (bf16)y;
    th[i] = yh;
    tl[i] = (bf16)(y - (float)yh);
}

// ---------------------------------------------------------------------------
// Kernel 1: pos[i] = dot(scores[i], targets[i]) in exact f32; zero neg_sum.
// ---------------------------------------------------------------------------
__global__ __launch_bounds__(256) void msl_pos(
    const float* __restrict__ s, const float* __restrict__ t,
    float* __restrict__ pos, float* __restrict__ neg)
{
    const int lane = threadIdx.x & 31;
    const int wave = threadIdx.x >> 5;
    const int row  = blockIdx.x * 8 + wave;
    const float* sr = s + (size_t)row * D_DIM;
    const float* tr = t + (size_t)row * D_DIM;
    float acc = 0.0f;
    #pragma unroll
    for (int j = 0; j < D_DIM / 32; ++j) {
        int k = lane + 32 * j;
        acc = fmaf(sr[k], tr[k], acc);
    }
    acc += __shfl_xor(acc, 1, 32);
    acc += __shfl_xor(acc, 2, 32);
    acc += __shfl_xor(acc, 4, 32);
    acc += __shfl_xor(acc, 8, 32);
    acc += __shfl_xor(acc, 16, 32);
    if (lane == 0) {
        pos[row] = acc;
        neg[row] = 0.0f;   // re-initialized every launch (graph-replay safe)
    }
}

// ---------------------------------------------------------------------------
// Fragment helpers: CDNA5 16-bit 16x32 WMMA lane layout (wave32):
//   lane<16 : its row, K = {kb+0..7, kb+16..23} with kb=0
//   lane>=16: same pattern with kb=8
// ---------------------------------------------------------------------------
__device__ __forceinline__ v16bf msl_load_frag(const bf16* __restrict__ p)
{
    v8bf a = *(const v8bf*)(p);
    v8bf b = *(const v8bf*)(p + 16);
    v16bf r;
    #pragma unroll
    for (int e = 0; e < 8; ++e) { r[e] = a[e]; r[e + 8] = b[e]; }
    return r;
}

__device__ __forceinline__ v8f msl_wmma(v16bf a, v16bf b, v8f c)
{
    return __builtin_amdgcn_wmma_f32_16x16x32_bf16(
        false, a, false, b, (short)0, c, false, false);
}

// Async DMA: global (16B) -> LDS, tracked by ASYNCcnt (gfx1250 path).
__device__ __forceinline__ void msl_async16(unsigned lds_off, const bf16* gptr)
{
    asm volatile("global_load_async_to_lds_b128 %0, %1, off"
                 :: "v"(lds_off), "v"(gptr) : "memory");
}
__device__ __forceinline__ void msl_wait_async0()
{
    asm volatile("s_wait_asynccnt 0" ::: "memory");
}

// ---------------------------------------------------------------------------
// Kernel 2: fused GEMM + masked-exp row reduction.
// Block = 256 threads = 8 waves. Block tile = 256 (M) x 64 (N).
// Wave w owns rows [m0+32w .. m0+32w+31] as two 16-row WMMA tiles x 4 N-tiles.
// B (targets hi/lo) k-chunk staged once per block into LDS via async DMA,
// double-buffered; A fragments stream from L2. bf16x3 => ~f32 accuracy.
// ---------------------------------------------------------------------------
__global__ __launch_bounds__(256) void msl_gemm(
    const bf16* __restrict__ sh, const bf16* __restrict__ sl,
    const bf16* __restrict__ th, const bf16* __restrict__ tl,
    const float* __restrict__ pos, float* __restrict__ neg)
{
    __shared__ bf16 sB[2][2][64 * 32];   // [buf][plane hi/lo][row][k] : 16 KB

    const int tid  = threadIdx.x;
    const int lane = tid & 31;
    const int wave = tid >> 5;
    const int m0   = blockIdx.x * 256;
    const int n0   = blockIdx.y * 64;
    const int mw   = m0 + wave * 32;
    const int rsel = lane & 15;
    const int hsel = (lane >> 4) & 1;
    const int kb   = hsel * 8;

    const unsigned ldsBase = (unsigned)(uintptr_t)&sB[0][0][0]; // LDS byte offset
    const unsigned planeB  = 64 * 32 * sizeof(bf16);            // 4096 B
    const unsigned bufB    = 2 * planeB;                        // 8192 B

    // Staging role of this thread: one 16B segment of each plane per chunk.
    const int srow  = tid >> 2;          // 0..63  (B row = sim column)
    const int spart = tid & 3;           // 0..3   (16B piece of 64B row-slice)
    const size_t gstage = (size_t)(n0 + srow) * D_DIM + spart * 8;
    const unsigned lstage = (unsigned)((srow * 32 + spart * 8) * sizeof(bf16));

    // Prologue: stage chunk 0 into buffer 0.
    msl_async16(ldsBase + lstage,          th + gstage);
    msl_async16(ldsBase + planeB + lstage, tl + gstage);

    v8f acc[2][4] = {};
    const size_t arow0 = (size_t)(mw + rsel) * D_DIM;
    const size_t arow1 = (size_t)(mw + 16 + rsel) * D_DIM;

    for (int kc = 0; kc < D_DIM / 32; ++kc) {
        const int cur = kc & 1;
        msl_wait_async0();        // this wave's DMA for buf[cur] complete
        __syncthreads();          // everyone's DMA visible; prev compute done

        if (kc < D_DIM / 32 - 1) {
            // Stage next chunk into the other buffer (safe: barrier above
            // guarantees all waves finished reading it two chunks ago).
            const size_t gnext = gstage + (size_t)(kc + 1) * 32;
            const unsigned lb = ldsBase + (unsigned)(cur ^ 1) * bufB + lstage;
            msl_async16(lb,          th + gnext);
            msl_async16(lb + planeB, tl + gnext);
        }

        const int k0 = kc * 32;
        v16bf aH0 = msl_load_frag(sh + arow0 + k0 + kb);
        v16bf aL0 = msl_load_frag(sl + arow0 + k0 + kb);
        v16bf aH1 = msl_load_frag(sh + arow1 + k0 + kb);
        v16bf aL1 = msl_load_frag(sl + arow1 + k0 + kb);

        #pragma unroll
        for (int t = 0; t < 4; ++t) {
            const int boff = (t * 16 + rsel) * 32 + kb;
            v16bf bH = msl_load_frag(&sB[cur][0][boff]);
            v16bf bL = msl_load_frag(&sB[cur][1][boff]);
            acc[0][t] = msl_wmma(aH0, bL, acc[0][t]);
            acc[0][t] = msl_wmma(aL0, bH, acc[0][t]);
            acc[0][t] = msl_wmma(aH0, bH, acc[0][t]);
            acc[1][t] = msl_wmma(aH1, bL, acc[1][t]);
            acc[1][t] = msl_wmma(aL1, bH, acc[1][t]);
            acc[1][t] = msl_wmma(aH1, bH, acc[1][t]);
        }
    }

    // Epilogue: C/D layout -> VGPR r, lane l holds
    // (row = mw + 16*i + r + 8*hsel, col = n0 + t*16 + (l&15)).
    #pragma unroll
    for (int i = 0; i < 2; ++i) {
        float p[8], rowsum[8];
        #pragma unroll
        for (int r = 0; r < 8; ++r) {
            p[r] = pos[mw + 16 * i + r + 8 * hsel];
            rowsum[r] = 0.0f;
        }
        #pragma unroll
        for (int t = 0; t < 4; ++t) {
            #pragma unroll
            for (int r = 0; r < 8; ++r) {
                float sv = acc[i][t][r];
                bool m = (fabsf(sv - p[r]) > MSL_EPS) && (sv + MARGIN > p[r]);
                rowsum[r] += m ? __expf(ALPHA * (sv - BASE)) : 0.0f;
            }
        }
        #pragma unroll
        for (int r = 0; r < 8; ++r) {
            float v = rowsum[r];
            v += __shfl_xor(v, 1, 32);
            v += __shfl_xor(v, 2, 32);
            v += __shfl_xor(v, 4, 32);
            v += __shfl_xor(v, 8, 32);
            if (rsel == 0)
                atomicAdd(neg + mw + 16 * i + r + 8 * hsel, v);
        }
    }
}

// ---------------------------------------------------------------------------
// Kernel 3: per-row loss terms + global mean (single block).
// ---------------------------------------------------------------------------
__global__ __launch_bounds__(256) void msl_final(
    const float* __restrict__ pos, const float* __restrict__ neg,
    float* __restrict__ out, int n)
{
    __shared__ float red[256];
    float acc = 0.0f;
    for (int i = threadIdx.x; i < n; i += 256) {
        float ns = neg[i];
        if (ns > 0.0f) {                                   // has_neg
            float pv = pos[i];
            float pos_loss = log1pf(__expf(-BETA * (pv - BASE))) / BETA;
            float neg_loss = log1pf(ns) / ALPHA;
            acc += pos_loss + neg_loss;
        }
    }
    red[threadIdx.x] = acc;
    __syncthreads();
    #pragma unroll
    for (int s = 128; s > 0; s >>= 1) {
        if (threadIdx.x < s) red[threadIdx.x] += red[threadIdx.x + s];
        __syncthreads();
    }
    if (threadIdx.x == 0) out[0] = red[0] / (float)n;
}

// ---------------------------------------------------------------------------
extern "C" void kernel_launch(void* const* d_in, const int* in_sizes, int n_in,
                              void* d_out, int out_size, void* d_ws, size_t ws_size,
                              hipStream_t stream)
{
    const float* scores  = (const float*)d_in[0];
    const float* targets = (const float*)d_in[1];
    float* out = (float*)d_out;

    const int total = in_sizes[0];        // n * d
    const int n = total / D_DIM;          // 8192

    const size_t plane = (size_t)total * sizeof(bf16);
    char* ws = (char*)d_ws;
    bf16* sh = (bf16*)(ws);
    bf16* sl = (bf16*)(ws + plane);
    bf16* th = (bf16*)(ws + 2 * plane);
    bf16* tl = (bf16*)(ws + 3 * plane);
    float* pos = (float*)(ws + 4 * plane);
    float* neg = pos + n;

    msl_convert<<<(total + 255) / 256, 256, 0, stream>>>(
        scores, targets, sh, sl, th, tl, total);

    msl_pos<<<n / 8, 256, 0, stream>>>(scores, targets, pos, neg);

    dim3 grid(n / 256, n / 64);           // 32 x 128 blocks
    msl_gemm<<<grid, 256, 0, stream>>>(sh, sl, th, tl, pos, neg);

    msl_final<<<1, 256, 0, stream>>>(pos, neg, out, n);
}
